// SelfAttentionBlock3D_37314675867976
// MI455X (gfx1250) — matrix operand order repaired and checked by reading, not compiled, not run on hardware
//
#include <hip/hip_runtime.h>

typedef __attribute__((ext_vector_type(16))) _Float16 v16h;
typedef __attribute__((ext_vector_type(8)))  _Float16 v8h;
typedef __attribute__((ext_vector_type(2)))  _Float16 v2h;
typedef __attribute__((ext_vector_type(8)))  float    v8f;

namespace {
constexpr int BB  = 2;
constexpr int CIN = 64;
constexpr int KC  = 32;
constexpr int VC  = 64;
constexpr int OC  = 64;
constexpr int SS  = 48;
constexpr int NN  = SS * SS * SS;   // 110592
constexpr int MM  = 756;
constexpr int MP  = 768;            // padded to 48 col-tiles
constexpr float GN_EPS = 1e-5f;
constexpr float SCALE  = 0.17677669529663687f; // 32^-0.5
}

__device__ __forceinline__ v8h ldv8h(const _Float16* p) {
  return *reinterpret_cast<const v8h*>(p);
}
__device__ __forceinline__ int v2h_to_i(v2h x) {
  union { v2h h; int i; } u; u.h = x; return u.i;
}
__device__ __forceinline__ v2h i_to_v2h(int x) {
  union { v2h h; int i; } u; u.i = x; return u.h;
}
__device__ __forceinline__ v2h hmax2(v2h a, v2h b) {
  v2h r;
  r[0] = (a[0] > b[0]) ? a[0] : b[0];
  r[1] = (a[1] > b[1]) ? a[1] : b[1];
  return r;
}

// ---------------------------------------------------------------- zero stats
__global__ __launch_bounds__(64) void k_zero_stats(float* gstats) {
  if (threadIdx.x < 64) gstats[threadIdx.x] = 0.f;
}

// ------------------------------------------------- cast weights to f16 once
__global__ __launch_bounds__(256) void k_wcast(
    const float* __restrict__ wk, const float* __restrict__ wv,
    const float* __restrict__ wo, _Float16* __restrict__ wkh,
    _Float16* __restrict__ wvh, _Float16* __restrict__ woh) {
  const int i = blockIdx.x * 256 + threadIdx.x;
  if (i < KC * CIN) wkh[i] = (_Float16)wk[i];
  if (i < VC * CIN) wvh[i] = (_Float16)wv[i];
  if (i < OC * VC)  woh[i] = (_Float16)wo[i];
}

// ------------------------- transpose+cast x: [b][c][n] f32 -> [b][n][c] f16
__global__ __launch_bounds__(256) void k_xcast(const float* __restrict__ x,
                                               _Float16* __restrict__ xh) {
  const int tid = blockIdx.x * 256 + threadIdx.x;
  const int b = tid / NN;
  const int n = tid - b * NN;
  const float* xp = x + (size_t)b * CIN * NN + n;
  _Float16* op = xh + ((size_t)b * NN + n) * CIN;
#pragma unroll
  for (int c = 0; c < CIN; ++c) op[c] = (_Float16)xp[(size_t)c * NN];
}

// -------------------- WMMA conv1x1: [16 pos x 64] x [64 x (32 key + 64 val)]
__global__ __launch_bounds__(32) void k_conv(
    const _Float16* __restrict__ xh, const _Float16* __restrict__ wkh,
    const _Float16* __restrict__ wvh, const float* __restrict__ bk,
    const float* __restrict__ bv, float* __restrict__ qraw,
    _Float16* __restrict__ vh) {
  const int nstrips = NN / 16;
  const int b = blockIdx.x / nstrips;
  const int strip = blockIdx.x - b * nstrips;
  const int n0 = strip * 16;
  const int lane = threadIdx.x;
  const int col = lane & 15;
  const int hi = lane >> 4;

  // A: x tile 16x64 as two 16x32 f16 A-matrices
  v16h aX0, aX1;
  {
    const _Float16* xp = xh + ((size_t)b * NN + n0 + col) * CIN;
    const v8h a = ldv8h(xp + hi * 16);
    const v8h c = ldv8h(xp + hi * 16 + 8);
    const v8h d = ldv8h(xp + 32 + hi * 16);
    const v8h e = ldv8h(xp + 32 + hi * 16 + 8);
#pragma unroll
    for (int i = 0; i < 8; ++i) {
      aX0[i] = a[i]; aX0[8 + i] = c[i];
      aX1[i] = d[i]; aX1[8 + i] = e[i];
    }
  }
  const v8f z = {};

  // key conv -> qraw (f32, channel-major)
#pragma unroll
  for (int t = 0; t < 2; ++t) {
    const int oc = t * 16 + col;
    const _Float16* wp = wkh + oc * CIN;
    v16h b0, b1;
    {
      const v8h l0 = ldv8h(wp + hi * 16);
      const v8h l1 = ldv8h(wp + hi * 16 + 8);
      const v8h l2 = ldv8h(wp + 32 + hi * 16);
      const v8h l3 = ldv8h(wp + 32 + hi * 16 + 8);
#pragma unroll
      for (int i = 0; i < 8; ++i) {
        b0[i] = l0[i]; b0[8 + i] = l1[i];
        b1[i] = l2[i]; b1[8 + i] = l3[i];
      }
    }
    v8f c = __builtin_amdgcn_wmma_f32_16x16x32_f16(false, aX0, false, b0, (short)0, z, false, false);
    c = __builtin_amdgcn_wmma_f32_16x16x32_f16(false, aX1, false, b1, (short)0, c, false, false);
    const float bo = bk[oc];
    float* qp = qraw + ((size_t)b * KC + oc) * NN + n0 + hi * 8;
#pragma unroll
    for (int v = 0; v < 8; ++v) qp[v] = c[v] + bo;
  }

  // value conv -> vh (f16, channel-major)
#pragma unroll
  for (int t = 0; t < 4; ++t) {
    const int oc = t * 16 + col;
    const _Float16* wp = wvh + oc * CIN;
    v16h b0, b1;
    {
      const v8h l0 = ldv8h(wp + hi * 16);
      const v8h l1 = ldv8h(wp + hi * 16 + 8);
      const v8h l2 = ldv8h(wp + 32 + hi * 16);
      const v8h l3 = ldv8h(wp + 32 + hi * 16 + 8);
#pragma unroll
      for (int i = 0; i < 8; ++i) {
        b0[i] = l0[i]; b0[8 + i] = l1[i];
        b1[i] = l2[i]; b1[8 + i] = l3[i];
      }
    }
    v8f c = __builtin_amdgcn_wmma_f32_16x16x32_f16(false, aX0, false, b0, (short)0, z, false, false);
    c = __builtin_amdgcn_wmma_f32_16x16x32_f16(false, aX1, false, b1, (short)0, c, false, false);
    const float bo = bv[oc];
    v8h o;
#pragma unroll
    for (int v = 0; v < 8; ++v) o[v] = (_Float16)(c[v] + bo);
    *reinterpret_cast<v8h*>(vh + ((size_t)b * VC + oc) * NN + n0 + hi * 8) = o;
  }
}

// -------------------------------- GN stats: reduce qraw per (batch, group)
__global__ __launch_bounds__(256) void k_stats(const float* __restrict__ qraw,
                                               float* __restrict__ gstats) {
  const int bg = blockIdx.x >> 6;     // 0..7 = b*4+g (8 contig channels each)
  const int chunk = blockIdx.x & 63;
  const int per = (8 * NN) / 64;      // 13824
  const float* p = qraw + (size_t)bg * 8 * NN + (size_t)chunk * per;
  float s = 0.f, q = 0.f;
  for (int i = threadIdx.x; i < per; i += 256) {
    const float v = p[i];
    s += v; q += v * v;
  }
  __shared__ float rs[256], rq[256];
  rs[threadIdx.x] = s; rq[threadIdx.x] = q;
  __syncthreads();
  for (int st = 128; st > 0; st >>= 1) {
    if (threadIdx.x < st) {
      rs[threadIdx.x] += rs[threadIdx.x + st];
      rq[threadIdx.x] += rq[threadIdx.x + st];
    }
    __syncthreads();
  }
  if (threadIdx.x == 0) {
    atomicAdd(&gstats[bg * 2], rs[0]);
    atomicAdd(&gstats[bg * 2 + 1], rq[0]);
  }
}

// ---------------------------------------------------------------- GN finalize
__global__ __launch_bounds__(32) void k_gn_final(float* gstats) {
  const int i = threadIdx.x;
  if (i < BB * 4) {
    const float cnt = (float)((KC / 4) * NN);
    const float mu  = gstats[i * 2] / cnt;
    const float var = gstats[i * 2 + 1] / cnt - mu * mu;
    gstats[16 + i * 2]     = mu;
    gstats[16 + i * 2 + 1] = rsqrtf(var + GN_EPS);
  }
}

// ------------- normalize + ReLU -> f16, transpose to position-major [n][32]
__global__ __launch_bounds__(256) void k_norm(
    const float* __restrict__ qraw, const float* __restrict__ gstats,
    const float* __restrict__ gw, const float* __restrict__ gb,
    _Float16* __restrict__ qh) {
  const int tid = blockIdx.x * 256 + threadIdx.x;
  const int b = tid / NN;
  const int n = tid - b * NN;
  const float* qp = qraw + (size_t)b * KC * NN + n;
  _Float16* op = qh + ((size_t)b * NN + n) * KC;
#pragma unroll
  for (int j = 0; j < KC; ++j) {
    const int g = j >> 3;
    const float mu = gstats[16 + ((b << 2) + g) * 2];
    const float rs = gstats[16 + ((b << 2) + g) * 2 + 1];
    const float y  = (qp[(size_t)j * NN] - mu) * rs * gw[j] + gb[j];
    op[j] = (_Float16)fmaxf(y, 0.f);
  }
}

// -------------------------------------------------------------- PSP decoding
__device__ __forceinline__ void decode_m(int m, int& s, int& bx, int& by, int& bz) {
  int mloc;
  if (m < 1)        { s = 1; mloc = m; }
  else if (m < 28)  { s = 3; mloc = m - 1; }
  else if (m < 244) { s = 6; mloc = m - 28; }
  else              { s = 8; mloc = m - 244; }
  bz = mloc / (s * s);
  const int r = mloc - bz * s * s;
  by = r / s;
  bx = r - by * s;
}

// ----------------------------------------------------- PSP pool of qk -> key_m
__global__ __launch_bounds__(256) void k_pool_key(const _Float16* __restrict__ qh,
                                                  _Float16* __restrict__ keym) {
  const int b = blockIdx.x / MP;
  const int m = blockIdx.x - b * MP;
  if (m >= MM) {
    if (threadIdx.x < KC) keym[((size_t)b * MP + m) * KC + threadIdx.x] = (_Float16)0.f;
    return;
  }
  int s, bx, by, bz;
  decode_m(m, s, bx, by, bz);
  const int Bs = SS / s;
  const int cnt = Bs * Bs * Bs;
  float acc[KC];
#pragma unroll
  for (int j = 0; j < KC; ++j) acc[j] = 0.f;
  for (int idx = threadIdx.x; idx < cnt; idx += 256) {
    const int lz = idx / (Bs * Bs);
    const int r  = idx - lz * Bs * Bs;
    const int ly = r / Bs;
    const int lx = r - ly * Bs;
    const int n  = ((bz * Bs + lz) * SS + by * Bs + ly) * SS + bx * Bs + lx;
    const _Float16* qp = qh + ((size_t)b * NN + n) * KC;
#pragma unroll
    for (int j = 0; j < KC; ++j) acc[j] += (float)qp[j];
  }
  __shared__ float red[256][KC + 1];
#pragma unroll
  for (int j = 0; j < KC; ++j) red[threadIdx.x][j] = acc[j];
  __syncthreads();
  for (int st = 128; st > 0; st >>= 1) {
    if (threadIdx.x < st) {
#pragma unroll
      for (int j = 0; j < KC; ++j) red[threadIdx.x][j] += red[threadIdx.x + st][j];
    }
    __syncthreads();
  }
  if (threadIdx.x < KC)
    keym[((size_t)b * MP + m) * KC + threadIdx.x] =
        (_Float16)(red[0][threadIdx.x] / (float)cnt);
}

// --------------------- PSP pool of value (channel-major) -> value_T[v][m]
__global__ __launch_bounds__(128) void k_pool_val(const _Float16* __restrict__ vh,
                                                  _Float16* __restrict__ valT) {
  const int b = blockIdx.x / MP;
  const int m = blockIdx.x - b * MP;
  if (m >= MM) {
    if (threadIdx.x < VC) valT[((size_t)b * VC + threadIdx.x) * MP + m] = (_Float16)0.f;
    return;
  }
  int s, bx, by, bz;
  decode_m(m, s, bx, by, bz);
  const int Bs = SS / s;
  const int cnt = Bs * Bs * Bs;
  float acc[VC];
#pragma unroll
  for (int j = 0; j < VC; ++j) acc[j] = 0.f;
  for (int idx = threadIdx.x; idx < cnt; idx += 128) {
    const int lz = idx / (Bs * Bs);
    const int r  = idx - lz * Bs * Bs;
    const int ly = r / Bs;
    const int lx = r - ly * Bs;
    const int n  = ((bz * Bs + lz) * SS + by * Bs + ly) * SS + bx * Bs + lx;
    const _Float16* vp = vh + (size_t)b * VC * NN + n;
#pragma unroll
    for (int j = 0; j < VC; ++j) acc[j] += (float)vp[(size_t)j * NN];
  }
  __shared__ float red[128][VC + 1];
#pragma unroll
  for (int j = 0; j < VC; ++j) red[threadIdx.x][j] = acc[j];
  __syncthreads();
  for (int st = 64; st > 0; st >>= 1) {
    if (threadIdx.x < st) {
#pragma unroll
      for (int j = 0; j < VC; ++j) red[threadIdx.x][j] += red[threadIdx.x + st][j];
    }
    __syncthreads();
  }
  if (threadIdx.x < VC)
    valT[((size_t)b * VC + threadIdx.x) * MP + m] =
        (_Float16)(red[0][threadIdx.x] / (float)cnt);
}

// -------- fused attention: QK^T -> online softmax -> PV -> out-conv (WMMA)
//          row sums via ones-matrix WMMA; row max via packed-f16 butterflies
__global__ __launch_bounds__(32) void k_attn(
    const _Float16* __restrict__ qh, const _Float16* __restrict__ keym,
    const _Float16* __restrict__ valT, const _Float16* __restrict__ woh,
    const float* __restrict__ b_out, float* __restrict__ out) {
  const int nstrips = NN / 16;
  const int b = blockIdx.x / nstrips;
  const int strip = blockIdx.x - b * nstrips;
  const int n0 = strip * 16;
  const int lane = threadIdx.x;
  const int col = lane & 15;   // C-layout column / A-layout row
  const int hi = lane >> 4;    // lane half-group

  // A-matrix: Q tile 16x32 f16
  v16h aQ;
  {
    const _Float16* qp = qh + ((size_t)b * NN + n0 + col) * KC;
    const v8h lo = ldv8h(qp + hi * 16);
    const v8h hv = ldv8h(qp + hi * 16 + 8);
#pragma unroll
    for (int i = 0; i < 8; ++i) { aQ[i] = lo[i]; aQ[8 + i] = hv[i]; }
  }

  v16h ones;
#pragma unroll
  for (int i = 0; i < 16; ++i) ones[i] = (_Float16)1.0f;

  v8f acc0 = {}, acc1 = {}, acc2 = {}, acc3 = {}, accS = {};
  float mrow[8];
#pragma unroll
  for (int v = 0; v < 8; ++v) mrow[v] = -3.0e38f;

  __shared__ _Float16 pbuf[16 * 64];   // P tiles repack C-layout -> A-layout
  __shared__ _Float16 cbuf[16 * 64];   // ctx repack for the out-conv GEMM

  for (int jp = 0; jp < MP / 64; ++jp) {
    const int m0 = jp * 64;
    const v8f z = {};
    v8f s0, s1, s2, s3;
#pragma unroll
    for (int t = 0; t < 4; ++t) {
      const _Float16* kp = keym + ((size_t)b * MP + m0 + t * 16 + col) * KC + hi * 16;
      const v8h lo = ldv8h(kp);
      const v8h hv = ldv8h(kp + 8);
      v16h bK;
#pragma unroll
      for (int i = 0; i < 8; ++i) { bK[i] = lo[i]; bK[8 + i] = hv[i]; }
      const v8f r = __builtin_amdgcn_wmma_f32_16x16x32_f16(false, aQ, false, bK, (short)0, z, false, false);
      if (t == 0) s0 = r; else if (t == 1) s1 = r; else if (t == 2) s2 = r; else s3 = r;
    }

    const bool pad3 = (jp == (MP / 64 - 1)) && (col >= (MM - (MP - 16)));  // col>=4 on last tile
#pragma unroll
    for (int v = 0; v < 8; ++v) {
      s0[v] *= SCALE;
      s1[v] *= SCALE;
      s2[v] *= SCALE;
      s3[v] = pad3 ? -3.0e38f : s3[v] * SCALE;
    }

    // per-row local max over 64 cols, then packed-f16 butterfly over 16 lanes
    v2h pk[4];
#pragma unroll
    for (int j = 0; j < 4; ++j) {
      const float a = fmaxf(fmaxf(s0[2 * j], s1[2 * j]), fmaxf(s2[2 * j], s3[2 * j]));
      const float c = fmaxf(fmaxf(s0[2 * j + 1], s1[2 * j + 1]), fmaxf(s2[2 * j + 1], s3[2 * j + 1]));
      v2h p; p[0] = (_Float16)a; p[1] = (_Float16)c;
      pk[j] = p;
    }
#pragma unroll
    for (int off = 1; off < 16; off <<= 1) {
#pragma unroll
      for (int j = 0; j < 4; ++j)
        pk[j] = hmax2(pk[j], i_to_v2h(__shfl_xor(v2h_to_i(pk[j]), off, 16)));
    }

    // online update: rescale accumulators (incl. the row-sum tile)
#pragma unroll
    for (int v = 0; v < 8; ++v) {
      const float rm = (float)pk[v >> 1][v & 1];
      const float mn = fmaxf(mrow[v], rm);
      const float co = __expf(mrow[v] - mn);
      mrow[v] = mn;
      acc0[v] *= co; acc1[v] *= co; acc2[v] *= co; acc3[v] *= co; accS[v] *= co;
    }

    // exp -> P tiles in LDS (A-layout repack)
    __syncthreads();
#pragma unroll
    for (int v = 0; v < 8; ++v) {
      const int row = v + hi * 8;
      pbuf[row * 64 + col]      = (_Float16)__expf(s0[v] - mrow[v]);
      pbuf[row * 64 + 16 + col] = (_Float16)__expf(s1[v] - mrow[v]);
      pbuf[row * 64 + 32 + col] = (_Float16)__expf(s2[v] - mrow[v]);
      pbuf[row * 64 + 48 + col] = (_Float16)__expf(s3[v] - mrow[v]);
    }
    __syncthreads();

    v16h aP0, aP1;
    {
      const _Float16* pp = pbuf + col * 64;
      const v8h a0 = ldv8h(pp + hi * 16);
      const v8h a1 = ldv8h(pp + hi * 16 + 8);
      const v8h a2 = ldv8h(pp + 32 + hi * 16);
      const v8h a3 = ldv8h(pp + 32 + hi * 16 + 8);
#pragma unroll
      for (int i = 0; i < 8; ++i) {
        aP0[i] = a0[i]; aP0[8 + i] = a1[i];
        aP1[i] = a2[i]; aP1[8 + i] = a3[i];
      }
    }

#pragma unroll
    for (int t = 0; t < 4; ++t) {
      const int vch = t * 16 + col;
      const _Float16* vp = valT + ((size_t)b * VC + vch) * MP + m0 + hi * 16;
      v16h bVa, bVb;
      {
        const v8h l0 = ldv8h(vp);
        const v8h l1 = ldv8h(vp + 8);
        const v8h l2 = ldv8h(vp + 32);
        const v8h l3 = ldv8h(vp + 40);
#pragma unroll
        for (int i = 0; i < 8; ++i) {
          bVa[i] = l0[i]; bVa[8 + i] = l1[i];
          bVb[i] = l2[i]; bVb[8 + i] = l3[i];
        }
      }
      if (t == 0) {
        acc0 = __builtin_amdgcn_wmma_f32_16x16x32_f16(false, aP0, false, bVa, (short)0, acc0, false, false);
        acc0 = __builtin_amdgcn_wmma_f32_16x16x32_f16(false, aP1, false, bVb, (short)0, acc0, false, false);
      } else if (t == 1) {
        acc1 = __builtin_amdgcn_wmma_f32_16x16x32_f16(false, aP0, false, bVa, (short)0, acc1, false, false);
        acc1 = __builtin_amdgcn_wmma_f32_16x16x32_f16(false, aP1, false, bVb, (short)0, acc1, false, false);
      } else if (t == 2) {
        acc2 = __builtin_amdgcn_wmma_f32_16x16x32_f16(false, aP0, false, bVa, (short)0, acc2, false, false);
        acc2 = __builtin_amdgcn_wmma_f32_16x16x32_f16(false, aP1, false, bVb, (short)0, acc2, false, false);
      } else {
        acc3 = __builtin_amdgcn_wmma_f32_16x16x32_f16(false, aP0, false, bVa, (short)0, acc3, false, false);
        acc3 = __builtin_amdgcn_wmma_f32_16x16x32_f16(false, aP1, false, bVb, (short)0, acc3, false, false);
      }
    }
    // row-sum recurrence: accS = accS*corr + rowsum(P)  (ones-matrix WMMA)
    accS = __builtin_amdgcn_wmma_f32_16x16x32_f16(false, aP0, false, ones, (short)0, accS, false, false);
    accS = __builtin_amdgcn_wmma_f32_16x16x32_f16(false, aP1, false, ones, (short)0, accS, false, false);
  }

  // finalize softmax normalization (every column of accS holds the row sum)
#pragma unroll
  for (int v = 0; v < 8; ++v) {
    const float inv = 1.f / accS[v];
    acc0[v] *= inv; acc1[v] *= inv; acc2[v] *= inv; acc3[v] *= inv;
  }

  // repack ctx into A-layout for the out-conv GEMM
  __syncthreads();
#pragma unroll
  for (int v = 0; v < 8; ++v) {
    const int row = v + hi * 8;
    cbuf[row * 64 + col]      = (_Float16)acc0[v];
    cbuf[row * 64 + 16 + col] = (_Float16)acc1[v];
    cbuf[row * 64 + 32 + col] = (_Float16)acc2[v];
    cbuf[row * 64 + 48 + col] = (_Float16)acc3[v];
  }
  __syncthreads();

  v16h aC0, aC1;
  {
    const _Float16* cp = cbuf + col * 64;
    const v8h a0 = ldv8h(cp + hi * 16);
    const v8h a1 = ldv8h(cp + hi * 16 + 8);
    const v8h a2 = ldv8h(cp + 32 + hi * 16);
    const v8h a3 = ldv8h(cp + 32 + hi * 16 + 8);
#pragma unroll
    for (int i = 0; i < 8; ++i) {
      aC0[i] = a0[i]; aC0[8 + i] = a1[i];
      aC1[i] = a2[i]; aC1[8 + i] = a3[i];
    }
  }

#pragma unroll
  for (int t = 0; t < 4; ++t) {
    const int oc = t * 16 + col;
    const _Float16* wp = woh + oc * VC;
    v16h bW0, bW1;
    {
      const v8h l0 = ldv8h(wp + hi * 16);
      const v8h l1 = ldv8h(wp + hi * 16 + 8);
      const v8h l2 = ldv8h(wp + 32 + hi * 16);
      const v8h l3 = ldv8h(wp + 32 + hi * 16 + 8);
#pragma unroll
      for (int i = 0; i < 8; ++i) {
        bW0[i] = l0[i]; bW0[8 + i] = l1[i];
        bW1[i] = l2[i]; bW1[8 + i] = l3[i];
      }
    }
    v8f o = {};
    o = __builtin_amdgcn_wmma_f32_16x16x32_f16(false, aC0, false, bW0, (short)0, o, false, false);
    o = __builtin_amdgcn_wmma_f32_16x16x32_f16(false, aC1, false, bW1, (short)0, o, false, false);
    const float bo = b_out[oc];
    float* op = out + ((size_t)b * OC + oc) * NN + n0 + hi * 8;
#pragma unroll
    for (int v = 0; v < 8; ++v) op[v] = o[v] + bo;
  }
}

// ------------------------------------------------------------------- launcher
extern "C" void kernel_launch(void* const* d_in, const int* in_sizes, int n_in,
                              void* d_out, int out_size, void* d_ws, size_t ws_size,
                              hipStream_t stream) {
  (void)in_sizes; (void)n_in; (void)out_size; (void)ws_size;
  const float* x     = (const float*)d_in[0];
  const float* w_key = (const float*)d_in[1];
  const float* b_key = (const float*)d_in[2];
  const float* gn_w  = (const float*)d_in[3];
  const float* gn_b  = (const float*)d_in[4];
  const float* w_val = (const float*)d_in[5];
  const float* b_val = (const float*)d_in[6];
  const float* w_out = (const float*)d_in[7];
  const float* b_out = (const float*)d_in[8];
  float* out = (float*)d_out;

  char* ws = (char*)d_ws;
  size_t off = 0;
  float* gstats  = (float*)(ws + off);    off += 256;
  _Float16* xh   = (_Float16*)(ws + off); off += (size_t)BB * NN * CIN * sizeof(_Float16);
  float* qraw    = (float*)(ws + off);    off += (size_t)BB * KC * NN * sizeof(float);
  _Float16* qh   = (_Float16*)(ws + off); off += (size_t)BB * NN * KC * sizeof(_Float16);
  _Float16* vh   = (_Float16*)(ws + off); off += (size_t)BB * VC * NN * sizeof(_Float16);
  _Float16* keym = (_Float16*)(ws + off); off += (size_t)BB * MP * KC * sizeof(_Float16);
  _Float16* valT = (_Float16*)(ws + off); off += (size_t)BB * VC * MP * sizeof(_Float16);
  _Float16* wkh  = (_Float16*)(ws + off); off += (size_t)KC * CIN * sizeof(_Float16);
  _Float16* wvh  = (_Float16*)(ws + off); off += (size_t)VC * CIN * sizeof(_Float16);
  _Float16* woh  = (_Float16*)(ws + off); off += (size_t)OC * VC * sizeof(_Float16);

  const int nposblk = BB * NN / 256;      // 864
  const int nstrips = BB * (NN / 16);     // 13824

  k_zero_stats<<<1, 64, 0, stream>>>(gstats);
  k_wcast<<<16, 256, 0, stream>>>(w_key, w_val, w_out, wkh, wvh, woh);
  k_xcast<<<nposblk, 256, 0, stream>>>(x, xh);
  k_conv<<<nstrips, 32, 0, stream>>>(xh, wkh, wvh, b_key, b_val, qraw, vh);
  k_stats<<<8 * 64, 256, 0, stream>>>(qraw, gstats);
  k_gn_final<<<1, 32, 0, stream>>>(gstats);
  k_norm<<<nposblk, 256, 0, stream>>>(qraw, gstats, gn_w, gn_b, qh);
  k_pool_key<<<BB * MP, 256, 0, stream>>>(qh, keym);
  k_pool_val<<<BB * MP, 128, 0, stream>>>(vh, valT);
  k_attn<<<nstrips, 32, 0, stream>>>(qh, keym, valT, woh, b_out, out);
}
